// MultiHeadSelfAttention_22402549416168
// MI455X (gfx1250) — matrix operand order repaired and checked by reading, not compiled, not run on hardware
//
#include <hip/hip_runtime.h>

// ---------------------------------------------------------------------------
// MHA for MI455X (gfx1250): bf16 WMMA GEMMs + flash attention with
// double-buffered async LDS staging (global_load_async_to_lds_b128 /
// s_wait_asynccnt) so operand fetch overlaps v_wmma issue.
// BS=2, S=2048, DIM=1024, H=16, DH=64.
// ---------------------------------------------------------------------------

#define BS  2
#define SEQ 2048
#define DIM 1024
#define NH  16
#define DH  64
#define MROWS (BS * SEQ)          // 4096

typedef __attribute__((ext_vector_type(16))) __bf16 v16bf;
typedef __attribute__((ext_vector_type(8)))  __bf16 v8bf;
typedef __attribute__((ext_vector_type(8)))  float  v8f;

// ---- helpers --------------------------------------------------------------

__device__ __forceinline__ unsigned short f32_to_bf16_rne(float f) {
    union { float f; unsigned int u; } v; v.f = f;
    unsigned int u = v.u;
    unsigned int r = u + 0x7FFFu + ((u >> 16) & 1u);   // round-to-nearest-even
    return (unsigned short)(r >> 16);
}

// A-operand (16x32 bf16, MxK): lane = half*16+row; vector = two contiguous
// 8-element chunks: K = [half*8 .. +7] and [16+half*8 .. +7].
__device__ __forceinline__ v16bf load_a_operand(const unsigned short* rowptr,
                                                int k0, int half) {
    v8bf lo = *reinterpret_cast<const v8bf*>(rowptr + k0 + half * 8);
    v8bf hi = *reinterpret_cast<const v8bf*>(rowptr + k0 + 16 + half * 8);
    return __builtin_shufflevector(lo, hi, 0, 1, 2, 3, 4, 5, 6, 7,
                                           8, 9, 10, 11, 12, 13, 14, 15);
}

// B-operand from a K-contiguous row in LDS/global: lane n = lane%16 holds
// K = [half*8 .. +7] and [16+half*8 .. +7]  (two aligned 16B chunks).
__device__ __forceinline__ v16bf load_b_operand16(const unsigned short* rowptr,
                                                  int half) {
    v8bf lo = *reinterpret_cast<const v8bf*>(rowptr + half * 8);
    v8bf hi = *reinterpret_cast<const v8bf*>(rowptr + 16 + half * 8);
    return __builtin_shufflevector(lo, hi, 0, 1, 2, 3, 4, 5, 6, 7,
                                           8, 9, 10, 11, 12, 13, 14, 15);
}

__device__ __forceinline__ v8f wmma_bf16(v16bf a, v16bf b, v8f c) {
    return __builtin_amdgcn_wmma_f32_16x16x32_bf16(
        false, a, false, b, (short)0, c, false, false);
}

// Per-lane async copy: 16B global -> LDS, tracked by ASYNCcnt.
__device__ __forceinline__ void async_copy_16B(const unsigned short* gptr,
                                               unsigned short* lptr) {
    unsigned lds = (unsigned)(uintptr_t)lptr;                 // LDS byte offset
    unsigned long long ga = (unsigned long long)(uintptr_t)gptr;
    asm volatile("global_load_async_to_lds_b128 %0, %1, off"
                 :: "v"(lds), "v"(ga) : "memory");
}

#define WAIT_ASYNC(n) asm volatile("s_wait_asynccnt " #n ::: "memory")

// ---- kernel 1: fp32 -> bf16 conversion ------------------------------------

__global__ __launch_bounds__(256) void convert_bf16(
    const float* __restrict__ in, unsigned short* __restrict__ out, int n) {
    int i = (blockIdx.x * blockDim.x + threadIdx.x) * 4;
    if (i + 3 < n) {
        float4 v = *reinterpret_cast<const float4*>(in + i);
        ushort4 o;
        o.x = f32_to_bf16_rne(v.x);
        o.y = f32_to_bf16_rne(v.y);
        o.z = f32_to_bf16_rne(v.z);
        o.w = f32_to_bf16_rne(v.w);
        *reinterpret_cast<ushort4*>(out + i) = o;
    }
}

// ---- kernel 2/4: GEMM  out[m][n] = (sum_k A[m][k] * W[n][k] + bias[n]) * sc
// Block = 256 thr (8 waves) = 128x64 C macro-tile, all waves share one
// 64-row W strip staged in LDS (double-buffered async), A register-pipelined.
// mode 0: Q proj -> bf16 (b,h,s,dh), scale 1/8    mode 1: K proj
// mode 2: V proj -> bf16 (b,h,dh,s) transposed    mode 3: O proj -> f32
// ---------------------------------------------------------------------------

#define WSTRIDE 40   // ushorts per W-tile row in LDS (80B, 16B aligned)

__global__ __launch_bounds__(256) void gemm_wmma(
    const unsigned short* __restrict__ A,
    const unsigned short* __restrict__ W,
    const float* __restrict__ bias,
    unsigned short* __restrict__ out_bf,
    float* __restrict__ out_f32,
    int mode) {
    __shared__ __attribute__((aligned(16))) unsigned short wlds[2][64 * WSTRIDE];

    const int tid  = threadIdx.x;
    const int lane = tid & 31;
    const int half = lane >> 4;
    const int lcol = lane & 15;
    const int wslot = tid >> 5;                 // 0..7
    const int mgroup = blockIdx.x >> 4;         // 0..31
    const int nstrip = blockIdx.x & 15;         // 0..15
    const int m0 = (mgroup * 8 + wslot) * 16;
    const int n0 = nstrip * 64;

    // async staging: thread -> chunk (row, 16B piece) of the 64x32 W tile
    const int wr = tid >> 2;                    // 0..63
    const int wc = tid & 3;                     // 0..3
    const unsigned short* wsrc = W + (size_t)(n0 + wr) * DIM + wc * 8;
    unsigned short* wdst0 = &wlds[0][wr * WSTRIDE + wc * 8];
    unsigned short* wdst1 = &wlds[1][wr * WSTRIDE + wc * 8];

    const unsigned short* arow = A + (size_t)(m0 + lcol) * DIM;

    async_copy_16B(wsrc, wdst0);                // prologue: k0 = 0 tile
    v16bf a_cur = load_a_operand(arow, 0, half);

    v8f acc[4] = {};
    for (int k0 = 0; k0 < DIM; k0 += 32) {
        const int buf = (k0 >> 5) & 1;
        __syncthreads();                        // WAR on buf^1
        if (k0 + 32 < DIM) {
            async_copy_16B(wsrc + (k0 + 32), buf ? wdst0 : wdst1);
            WAIT_ASYNC(0x1);                    // current buffer's load done
        } else {
            WAIT_ASYNC(0x0);
        }
        __syncthreads();                        // tile visible to all waves

        v16bf a_next = a_cur;
        if (k0 + 32 < DIM) a_next = load_a_operand(arow, k0 + 32, half);

        const unsigned short* wt = &wlds[buf][0];
#pragma unroll
        for (int t = 0; t < 4; ++t) {
            v16bf b = load_b_operand16(wt + (t * 16 + lcol) * WSTRIDE, half);
            acc[t] = wmma_bf16(a_cur, b, acc[t]);
        }
        a_cur = a_next;
    }

    const float scale = (mode == 0) ? 0.125f : 1.0f;  // 1/sqrt(64)
#pragma unroll
    for (int t = 0; t < 4; ++t) {
        const int n = n0 + t * 16 + lcol;
        const float bv = bias[n];
#pragma unroll
        for (int r = 0; r < 8; ++r) {
            const int m = m0 + r + 8 * half;
            const float val = (acc[t][r] + bv) * scale;
            if (mode == 3) {
                out_f32[(size_t)m * DIM + n] = val;
            } else {
                const int b = m >> 11, s = m & (SEQ - 1);
                const int h = n >> 6, dh = n & (DH - 1);
                size_t idx;
                if (mode == 2)
                    idx = (((size_t)(b * NH + h) * DH + dh) * SEQ) + s;
                else
                    idx = (((size_t)(b * NH + h) * SEQ + s) * DH) + dh;
                out_bf[idx] = f32_to_bf16_rne(val);
            }
        }
    }
}

// ---- kernel 3: flash attention --------------------------------------------
// Block = 4 waves = 4 q-tiles of the SAME (b,h).  Per j-block of 32 keys the
// K tile (32x64) and V^T tile (64x32) are shared by all 4 waves, staged in
// LDS via double-buffered async copies.  Softmax row stats via shuffle
// reductions; P goes D-layout -> LDS -> A-operand layout.
// ---------------------------------------------------------------------------

#define PSTRIDE 40   // P rows:  32 elems + pad  (80B)
#define KSTRIDE 72   // K rows:  64 elems + pad  (144B)
#define VSTRIDE 40   // V rows:  32 elems + pad  (80B)

__global__ __launch_bounds__(128) void flash_attn(
    const unsigned short* __restrict__ Q,   // (B,H,S,DH) bf16, pre-scaled
    const unsigned short* __restrict__ K,   // (B,H,S,DH) bf16
    const unsigned short* __restrict__ VT,  // (B,H,DH,S) bf16
    const int* __restrict__ mask,           // (B,S)
    unsigned short* __restrict__ CTX) {     // (B,S,DIM) bf16
    __shared__ __attribute__((aligned(16))) unsigned short klds[2][32 * KSTRIDE];
    __shared__ __attribute__((aligned(16))) unsigned short vlds[2][64 * VSTRIDE];
    __shared__ __attribute__((aligned(16))) unsigned short plds_all[4][16 * PSTRIDE];

    const int tid  = threadIdx.x;
    const int lane = tid & 31;
    const int half = lane >> 4;
    const int lcol = lane & 15;
    const int wslot = tid >> 5;
    unsigned short* plds = plds_all[wslot];

    const int bh = blockIdx.x >> 5;            // 0..31  (b*16+h)
    const int qgroup = blockIdx.x & 31;        // 32 groups of 4 q-tiles
    const int b = bh >> 4;
    const int q0 = (qgroup * 4 + wslot) * 16;

    const unsigned short* qbase = Q + (size_t)bh * SEQ * DH;
    const unsigned short* kbase = K + (size_t)bh * SEQ * DH;
    const unsigned short* vbase = VT + (size_t)bh * DH * SEQ;
    const int* mrow = mask + b * SEQ;

    // q A-operands for this wave's tile, reused across the whole loop
    const unsigned short* qrow = qbase + (size_t)(q0 + lcol) * DH;
    const v16bf qa0 = load_a_operand(qrow, 0, half);
    const v16bf qa1 = load_a_operand(qrow, 32, half);

    // async staging: 4 x 16B chunks per thread per j-block (2 K + 2 V)
    auto issue_tiles = [&](int buf, int j0) {
#pragma unroll
        for (int q = 0; q < 2; ++q) {
            const int id = tid + q * 128;          // 0..255
            const int r = id >> 3, c = id & 7;     // K: 32 rows x 8 chunks
            async_copy_16B(kbase + (size_t)(j0 + r) * DH + c * 8,
                           &klds[buf][r * KSTRIDE + c * 8]);
        }
#pragma unroll
        for (int q = 0; q < 2; ++q) {
            const int id = tid + q * 128;
            const int r = id >> 2, c = id & 3;     // V: 64 rows x 4 chunks
            async_copy_16B(vbase + (size_t)r * SEQ + j0 + c * 8,
                           &vlds[buf][r * VSTRIDE + c * 8]);
        }
    };

    float mrun[8], lrun[8], fac[8];
    v8f ctx[4] = {};
#pragma unroll
    for (int r = 0; r < 8; ++r) { mrun[r] = -3.0e38f; lrun[r] = 0.0f; }

    issue_tiles(0, 0);                             // prologue
    for (int j0 = 0; j0 < SEQ; j0 += 32) {
        const int buf = (j0 >> 5) & 1;
        __syncthreads();                           // WAR on buf^1
        if (j0 + 32 < SEQ) {
            issue_tiles(buf ^ 1, j0 + 32);
            WAIT_ASYNC(0x4);                       // current buffer complete
        } else {
            WAIT_ASYNC(0x0);
        }
        __syncthreads();                           // tiles visible block-wide

        // ---- scores from staged K tile ----
        const unsigned short* kt = &klds[buf][0];
        v8f s0 = {}, s1 = {};
        s0 = wmma_bf16(qa0, load_b_operand16(kt + lcol * KSTRIDE, half), s0);
        s0 = wmma_bf16(qa1, load_b_operand16(kt + lcol * KSTRIDE + 32, half), s0);
        s1 = wmma_bf16(qa0, load_b_operand16(kt + (16 + lcol) * KSTRIDE, half), s1);
        s1 = wmma_bf16(qa1, load_b_operand16(kt + (16 + lcol) * KSTRIDE + 32, half), s1);

        const int mk0 = mrow[j0 + lcol];
        const int mk1 = mrow[j0 + 16 + lcol];

        // ---- online softmax per row r (row index = r + 8*half) ----
#pragma unroll
        for (int r = 0; r < 8; ++r) {
            float a0 = mk0 ? s0[r] : -1.0e30f;
            float a1 = mk1 ? s1[r] : -1.0e30f;
            float mx = fmaxf(a0, a1);
#pragma unroll
            for (int off = 8; off; off >>= 1)
                mx = fmaxf(mx, __shfl_xor(mx, off, 32));
            const float mnew = fmaxf(mrun[r], mx);
            fac[r] = __expf(mrun[r] - mnew);
            mrun[r] = mnew;
            const float p0 = __expf(a0 - mnew);
            const float p1 = __expf(a1 - mnew);
            float rs = p0 + p1;
#pragma unroll
            for (int off = 8; off; off >>= 1)
                rs += __shfl_xor(rs, off, 32);
            lrun[r] = lrun[r] * fac[r] + rs;
            const int m = r + 8 * half;
            plds[m * PSTRIDE + lcol]      = f32_to_bf16_rne(p0);
            plds[m * PSTRIDE + 16 + lcol] = f32_to_bf16_rne(p1);
        }

#pragma unroll
        for (int t = 0; t < 4; ++t)
#pragma unroll
            for (int r = 0; r < 8; ++r) ctx[t][r] *= fac[r];

        // P: D-layout (LDS) -> A-operand layout
        v8bf plo = *reinterpret_cast<const v8bf*>(plds + lcol * PSTRIDE + half * 8);
        v8bf phi = *reinterpret_cast<const v8bf*>(plds + lcol * PSTRIDE + 16 + half * 8);
        v16bf pa = __builtin_shufflevector(plo, phi, 0, 1, 2, 3, 4, 5, 6, 7,
                                                     8, 9, 10, 11, 12, 13, 14, 15);

        // ---- context += P * V from staged V^T tile ----
        const unsigned short* vt = &vlds[buf][0];
#pragma unroll
        for (int t = 0; t < 4; ++t) {
            v16bf bv = load_b_operand16(vt + (t * 16 + lcol) * VSTRIDE, half);
            ctx[t] = wmma_bf16(pa, bv, ctx[t]);
        }
    }

    // ---- finalize & store (b, s, h*DH + dh) as bf16 ----
    const int h = bh & (NH - 1);
#pragma unroll
    for (int t = 0; t < 4; ++t) {
#pragma unroll
        for (int r = 0; r < 8; ++r) {
            const int s = q0 + r + 8 * half;
            const float val = ctx[t][r] / lrun[r];
            const size_t idx = ((size_t)b * SEQ + s) * DIM + h * DH + t * 16 + lcol;
            CTX[idx] = f32_to_bf16_rne(val);
        }
    }
}

// ---- host glue ------------------------------------------------------------

extern "C" void kernel_launch(void* const* d_in, const int* in_sizes, int n_in,
                              void* d_out, int out_size, void* d_ws, size_t ws_size,
                              hipStream_t stream) {
    const float* q_in = (const float*)d_in[0];
    const float* k_in = (const float*)d_in[1];
    const float* v_in = (const float*)d_in[2];
    const int*   mask = (const int*)d_in[3];
    const float* q_w  = (const float*)d_in[4];
    const float* q_b  = (const float*)d_in[5];
    const float* k_w  = (const float*)d_in[6];
    const float* k_b  = (const float*)d_in[7];
    const float* v_w  = (const float*)d_in[8];
    const float* v_b  = (const float*)d_in[9];
    const float* o_w  = (const float*)d_in[10];
    const float* o_b  = (const float*)d_in[11];
    float* out = (float*)d_out;

    const size_t NX = (size_t)MROWS * DIM;   // 4,194,304 activations
    const size_t NW = (size_t)DIM * DIM;     // 1,048,576 weights

    char* ws = (char*)d_ws;
    unsigned short* xq = (unsigned short*)(ws);
    unsigned short* xk = (unsigned short*)(ws + 8ull  * 1024 * 1024);
    unsigned short* xv = (unsigned short*)(ws + 16ull * 1024 * 1024);
    unsigned short* wq = (unsigned short*)(ws + 24ull * 1024 * 1024);
    unsigned short* wk = (unsigned short*)(ws + 26ull * 1024 * 1024);
    unsigned short* wv = (unsigned short*)(ws + 28ull * 1024 * 1024);
    unsigned short* wo = (unsigned short*)(ws + 30ull * 1024 * 1024);
    unsigned short* qp = (unsigned short*)(ws + 32ull * 1024 * 1024);
    unsigned short* kp = (unsigned short*)(ws + 40ull * 1024 * 1024);
    unsigned short* vt = (unsigned short*)(ws + 48ull * 1024 * 1024);
    unsigned short* cx = (unsigned short*)(ws + 56ull * 1024 * 1024);

    const int convT = 256;
    convert_bf16<<<(int)(NX / 4 / convT), convT, 0, stream>>>(q_in, xq, (int)NX);
    convert_bf16<<<(int)(NX / 4 / convT), convT, 0, stream>>>(k_in, xk, (int)NX);
    convert_bf16<<<(int)(NX / 4 / convT), convT, 0, stream>>>(v_in, xv, (int)NX);
    convert_bf16<<<(int)(NW / 4 / convT), convT, 0, stream>>>(q_w, wq, (int)NW);
    convert_bf16<<<(int)(NW / 4 / convT), convT, 0, stream>>>(k_w, wk, (int)NW);
    convert_bf16<<<(int)(NW / 4 / convT), convT, 0, stream>>>(v_w, wv, (int)NW);
    convert_bf16<<<(int)(NW / 4 / convT), convT, 0, stream>>>(o_w, wo, (int)NW);

    // 512 blocks x 256 thr: 32 m-groups (128 rows) x 16 n-strips (64 cols)
    gemm_wmma<<<512, 256, 0, stream>>>(xq, wq, q_b, qp, nullptr, 0);
    gemm_wmma<<<512, 256, 0, stream>>>(xk, wk, k_b, kp, nullptr, 1);
    gemm_wmma<<<512, 256, 0, stream>>>(xv, wv, v_b, vt, nullptr, 2);

    // 1024 blocks x 128 thr: (b,h) x 32 q-groups, 4 q-tiles/block share K/V
    flash_attn<<<1024, 128, 0, stream>>>(qp, kp, vt, mask, cx);

    gemm_wmma<<<512, 256, 0, stream>>>(cx, wo, o_b, nullptr, out, 3);
}
